// EmRNN_61950608278071
// MI455X (gfx1250) — compile-verified, matrix operand order checked by reading, not run on hardware
//
#include <hip/hip_runtime.h>
#include <hip/hip_bf16.h>
#include <math.h>

// ---------------- problem constants ----------------
static constexpr int kCELL   = 10;
static constexpr int kNSLOTS = 5;
static constexpr int kNREADS = 2;
static constexpr int kREAD   = kNREADS * kCELL;   // 20
static constexpr int kBS     = 256;
static constexpr int kT      = 512;
static constexpr int kD      = 128;
static constexpr int kH      = 128;
static constexpr int kCIN    = kD + kREAD;        // 148
static constexpr int kCINP   = 160;               // padded to multiple of 32
static constexpr int kCOUT   = kREAD + kCELL;     // 30
static constexpr int kCOUTP  = 32;
static constexpr int kG3H    = 3 * kH;            // 384
static constexpr int kOUTW   = kH + kH + kREAD;   // 276
static constexpr float kTAU  = 0.3f;
static constexpr float kCLIP = 20.0f;

// bf16 weight workspace layout (element offsets)
static constexpr size_t kOFF_WIH0 = 0;
static constexpr size_t kOFF_WHH0 = kOFF_WIH0 + (size_t)kG3H * kCINP;
static constexpr size_t kOFF_WIH1 = kOFF_WHH0 + (size_t)kG3H * kH;
static constexpr size_t kOFF_WHH1 = kOFF_WIH1 + (size_t)kG3H * kH;
static constexpr size_t kOFF_WIF  = kOFF_WHH1 + (size_t)kG3H * kH;
static constexpr size_t kWS_ELEMS = kOFF_WIF + (size_t)kCOUTP * kH;   // 212,992 elems = 416 KB

typedef __attribute__((ext_vector_type(16))) __bf16 v16bf;
typedef __attribute__((ext_vector_type(8)))  __bf16 v8bf;
typedef __attribute__((ext_vector_type(8)))  float  v8f;

__device__ __forceinline__ unsigned short f2bf(float f) {
  union { float f; unsigned u; } v; v.f = f;
  unsigned u = v.u;
  unsigned r = u + 0x7FFFu + ((u >> 16) & 1u);   // round-to-nearest-even
  return (unsigned short)(r >> 16);
}

// Load one 16x32 bf16 fragment (A or B^T view), row-major source with leading dim `ld`.
// 16-bit fragment striping: lanes 0-15 hold K chunks {k0..k0+7, k0+16..k0+23},
// lanes 16-31 hold {k0+8..k0+15, k0+24..k0+31} (two 16B contiguous loads per lane).
__device__ __forceinline__ v16bf load_frag(const unsigned short* base, int ld, int k0, int lane) {
  const int m  = lane & 15;
  const int hi = (lane >> 4) & 1;
  const unsigned short* p = base + (size_t)m * ld + k0 + hi * 8;
  union { v16bf v; v8bf h[2]; } f;
  f.h[0] = *reinterpret_cast<const v8bf*>(p);
  f.h[1] = *reinterpret_cast<const v8bf*>(p + 16);
  return f.v;
}

__device__ __forceinline__ void gemm_tile(const unsigned short* A, int lda,
                                          const unsigned short* B, int ldb,
                                          int ktiles, int lane, v8f& acc) {
#pragma unroll
  for (int kt = 0; kt < ktiles; ++kt) {
    v16bf a = load_frag(A, lda, kt * 32, lane);
    v16bf b = load_frag(B, ldb, kt * 32, lane);
    acc = __builtin_amdgcn_wmma_f32_16x16x32_bf16(
        /*neg_a=*/false, a, /*neg_b=*/false, b,
        /*c_mod=*/(short)0, acc, /*reuse_a=*/false, /*reuse_b=*/false);
  }
}

// C/D layout: VGPR r -> M = r + 8*(lane>=16), N = ntile*16 + (lane&15)
__device__ __forceinline__ void store_tile(float* C, int ldc, int nt, int lane, const v8f& acc) {
  const int n  = nt * 16 + (lane & 15);
  const int mb = (lane >> 4) * 8;
#pragma unroll
  for (int r = 0; r < 8; ++r) C[(size_t)(r + mb) * ldc + n] = acc[r];
}

// ---------------- weight prepack: f32 -> padded bf16 ----------------
__global__ void emrnn_prepack(const float* __restrict__ W_ih0, const float* __restrict__ W_hh0,
                              const float* __restrict__ W_ih1, const float* __restrict__ W_hh1,
                              const float* __restrict__ W_if, unsigned short* __restrict__ dst) {
  size_t i = (size_t)blockIdx.x * blockDim.x + threadIdx.x;
  if (i >= kWS_ELEMS) return;
  float v;
  if (i < kOFF_WHH0) {
    int r = (int)(i / kCINP), c = (int)(i % kCINP);
    v = (c < kCIN) ? W_ih0[(size_t)r * kCIN + c] : 0.f;
  } else if (i < kOFF_WIH1) {
    v = W_hh0[i - kOFF_WHH0];
  } else if (i < kOFF_WHH1) {
    v = W_ih1[i - kOFF_WIH1];
  } else if (i < kOFF_WIF) {
    v = W_hh1[i - kOFF_WHH1];
  } else {
    size_t j = i - kOFF_WIF;
    int r = (int)(j / kH), c = (int)(j % kH);
    v = (r < kCOUT) ? W_if[(size_t)r * kH + c] : 0.f;
  }
  dst[i] = f2bf(v);
}

// ---------------- persistent recurrent kernel ----------------
// Grid: BS/16 workgroups; block: 256 threads (8 wave32). Each WG owns 16 batch rows.
__global__ __launch_bounds__(256)
void emrnn_kernel(const float* __restrict__ x, const float* __restrict__ h0,
                  const float* __restrict__ b_ih0, const float* __restrict__ b_hh0,
                  const float* __restrict__ b_ih1, const float* __restrict__ b_hh1,
                  const float* __restrict__ b_if,
                  const unsigned short* __restrict__ wsb,
                  float* __restrict__ out) {
  __shared__ __align__(16) float h1s[16][kH];
  __shared__ __align__(16) float h2s[16][kH];
  __shared__ __align__(16) unsigned short ci_bf[16][kCINP];
  __shared__ __align__(16) unsigned short h1bf[16][kH];
  __shared__ __align__(16) unsigned short h2bf[16][kH];
  __shared__ __align__(16) float gis[16][kG3H];
  __shared__ __align__(16) float ghs[16][kG3H];
  __shared__ float mems[16][kCELL * kNSLOTS];   // [c*NSLOTS + n]
  __shared__ float lreads[16][kREAD];
  __shared__ float outs[16][kCOUTP];

  const int tid  = threadIdx.x;
  const int lane = tid & 31;
  const int wv   = tid >> 5;
  const int bg   = blockIdx.x * 16;
  const int row  = tid >> 4;          // 0..15
  const int c0   = (tid & 15) * 8;    // 0..120

  // ---- init: state from h0, memory/read zeroed ----
  for (int j = 0; j < 8; ++j) {
    int c = c0 + j;
    h1s[row][c] = h0[(size_t)(bg + row) * kH + c];
    h2s[row][c] = h0[(size_t)(kBS + bg + row) * kH + c];
  }
  if (tid < 16) {
    for (int i = 0; i < kCELL * kNSLOTS; ++i) mems[tid][i] = 0.f;
    for (int i = 0; i < kREAD; ++i) lreads[tid][i] = 0.f;
  }
  __syncthreads();

  for (int t = 0; t < kT; ++t) {
    // ---- P1: build bf16 operands: ci = [x_t | lread | 0pad], h1, h2 ----
    {
      const float* xr = x + ((size_t)(bg + row) * kT + t) * kD;
      for (int j = 0; j < 8; ++j) {
        int c = c0 + j;
        ci_bf[row][c] = f2bf(xr[c]);
        h1bf[row][c]  = f2bf(h1s[row][c]);
        h2bf[row][c]  = f2bf(h2s[row][c]);
      }
      for (int j = 0; j < 2; ++j) {                 // cols 128..159 (16x32 = 512 elems)
        int idx = tid * 2 + j;
        int r2  = idx >> 5;
        int c   = kD + (idx & 31);
        float v = (c < kCIN) ? lreads[r2][c - kD] : 0.f;
        ci_bf[r2][c] = f2bf(v);
      }
      if (t + 1 < kT && tid < 64) {                 // prefetch next x tile
        int r3 = tid >> 2, part = tid & 3;
        __builtin_prefetch(x + ((size_t)(bg + r3) * kT + (t + 1)) * kD + part * 32, 0, 3);
      }
    }
    __syncthreads();

    // ---- P2: layer-0 GEMMs: gi = ci @ Wih0^T, gh = h1 @ Whh0^T (24 tiles each) ----
    for (int i = 0; i < 3; ++i) {
      int nt = wv + i * 8;
      v8f a0 = {};
      gemm_tile(&ci_bf[0][0], kCINP, wsb + kOFF_WIH0 + (size_t)nt * 16 * kCINP, kCINP,
                kCINP / 32, lane, a0);
      store_tile(&gis[0][0], kG3H, nt, lane, a0);
      v8f a1 = {};
      gemm_tile(&h1bf[0][0], kH, wsb + kOFF_WHH0 + (size_t)nt * 16 * kH, kH,
                kH / 32, lane, a1);
      store_tile(&ghs[0][0], kG3H, nt, lane, a1);
    }
    __syncthreads();

    // ---- P3: GRU0 elementwise -> h1n ----
    {
      float* orow = out + ((size_t)(bg + row) * kT + t) * kOUTW;
      for (int j = 0; j < 8; ++j) {
        int c = c0 + j;
        float ir  = gis[row][c]          + b_ih0[c];
        float iz  = gis[row][kH + c]     + b_ih0[kH + c];
        float in_ = gis[row][2 * kH + c] + b_ih0[2 * kH + c];
        float hr  = ghs[row][c]          + b_hh0[c];
        float hz  = ghs[row][kH + c]     + b_hh0[kH + c];
        float hn  = ghs[row][2 * kH + c] + b_hh0[2 * kH + c];
        float r = 1.f / (1.f + __expf(-(ir + hr)));
        float z = 1.f / (1.f + __expf(-(iz + hz)));
        float n = tanhf(in_ + r * hn);
        float h = (1.f - z) * n + z * h1s[row][c];
        h1s[row][c]  = h;
        h1bf[row][c] = f2bf(h);
        orow[c] = h;
      }
    }
    __syncthreads();

    // ---- P4: layer-1 GEMMs: gi1 = h1n @ Wih1^T, gh1 = h2 @ Whh1^T ----
    for (int i = 0; i < 3; ++i) {
      int nt = wv + i * 8;
      v8f a0 = {};
      gemm_tile(&h1bf[0][0], kH, wsb + kOFF_WIH1 + (size_t)nt * 16 * kH, kH,
                kH / 32, lane, a0);
      store_tile(&gis[0][0], kG3H, nt, lane, a0);
      v8f a1 = {};
      gemm_tile(&h2bf[0][0], kH, wsb + kOFF_WHH1 + (size_t)nt * 16 * kH, kH,
                kH / 32, lane, a1);
      store_tile(&ghs[0][0], kG3H, nt, lane, a1);
    }
    __syncthreads();

    // ---- P5: GRU1 elementwise -> h2n ----
    {
      float* orow = out + ((size_t)(bg + row) * kT + t) * kOUTW;
      for (int j = 0; j < 8; ++j) {
        int c = c0 + j;
        float ir  = gis[row][c]          + b_ih1[c];
        float iz  = gis[row][kH + c]     + b_ih1[kH + c];
        float in_ = gis[row][2 * kH + c] + b_ih1[2 * kH + c];
        float hr  = ghs[row][c]          + b_hh1[c];
        float hz  = ghs[row][kH + c]     + b_hh1[kH + c];
        float hn  = ghs[row][2 * kH + c] + b_hh1[2 * kH + c];
        float r = 1.f / (1.f + __expf(-(ir + hr)));
        float z = 1.f / (1.f + __expf(-(iz + hz)));
        float n = tanhf(in_ + r * hn);
        float h = (1.f - z) * n + z * h2s[row][c];
        h2s[row][c]  = h;
        h2bf[row][c] = f2bf(h);
        orow[kH + c] = h;
      }
    }
    __syncthreads();

    // ---- P6: projection out = clip(relu(h2n @ Wif^T + b_if)) (2 tiles, waves 0-1) ----
    if (wv < 2) {
      v8f a0 = {};
      gemm_tile(&h2bf[0][0], kH, wsb + kOFF_WIF + (size_t)wv * 16 * kH, kH,
                kH / 32, lane, a0);
      int n  = wv * 16 + (lane & 15);
      int mb = (lane >> 4) * 8;
#pragma unroll
      for (int r = 0; r < 8; ++r) {
        float v = a0[r] + ((n < kCOUT) ? b_if[n] : 0.f);
        v = fminf(fmaxf(v, 0.f), kCLIP);
        outs[r + mb][n] = v;
      }
    }
    __syncthreads();

    // ---- P7: per-row memory write + cosine attention read ----
    if (tid < 16) {
      const int rr = tid;
      float* M = mems[rr];
      const int slot = t % kNSLOTS;
      for (int c = 0; c < kCELL; ++c) M[c * kNSLOTS + slot] = outs[rr][kREAD + c];

      float kn[kNREADS][kCELL];
      for (int r = 0; r < kNREADS; ++r) {
        float s = 0.f;
        for (int c = 0; c < kCELL; ++c) { float k = outs[rr][r * kCELL + c]; s += k * k; }
        float inv = 1.f / (sqrtf(s) + 1e-8f);
        for (int c = 0; c < kCELL; ++c) kn[r][c] = outs[rr][r * kCELL + c] * inv;
      }
      float minv[kNSLOTS];
      for (int nn = 0; nn < kNSLOTS; ++nn) {
        float s = 0.f;
        for (int c = 0; c < kCELL; ++c) { float v = M[c * kNSLOTS + nn]; s += v * v; }
        minv[nn] = 1.f / (sqrtf(s) + 1e-8f);
      }
      float* orow = out + ((size_t)(bg + rr) * kT + t) * kOUTW + 2 * kH;
      for (int r = 0; r < kNREADS; ++r) {
        float w[kNSLOTS];
        float mx = -1e30f;
        for (int nn = 0; nn < kNSLOTS; ++nn) {
          float s = 0.f;
          for (int c = 0; c < kCELL; ++c) s += kn[r][c] * M[c * kNSLOTS + nn];
          s = s * minv[nn] * (1.0f / kTAU);
          w[nn] = s;
          mx = fmaxf(mx, s);
        }
        float den = 0.f;
        for (int nn = 0; nn < kNSLOTS; ++nn) { w[nn] = expf(w[nn] - mx); den += w[nn]; }
        float iden = 1.f / den;
        for (int c = 0; c < kCELL; ++c) {
          float rd = 0.f;
          for (int nn = 0; nn < kNSLOTS; ++nn) rd += (w[nn] * iden) * M[c * kNSLOTS + nn];
          lreads[rr][r * kCELL + c] = rd;
          orow[r * kCELL + c] = rd;
        }
      }
    }
    __syncthreads();
  }
}

extern "C" void kernel_launch(void* const* d_in, const int* in_sizes, int n_in,
                              void* d_out, int out_size, void* d_ws, size_t ws_size,
                              hipStream_t stream) {
  const float* x     = (const float*)d_in[0];
  const float* h0    = (const float*)d_in[1];
  // d_in[2] = mem0 (all zeros) -- state initialized in-kernel
  const float* W_ih0 = (const float*)d_in[3];
  const float* W_hh0 = (const float*)d_in[4];
  const float* b_ih0 = (const float*)d_in[5];
  const float* b_hh0 = (const float*)d_in[6];
  const float* W_ih1 = (const float*)d_in[7];
  const float* W_hh1 = (const float*)d_in[8];
  const float* b_ih1 = (const float*)d_in[9];
  const float* b_hh1 = (const float*)d_in[10];
  const float* W_if  = (const float*)d_in[11];
  const float* b_if  = (const float*)d_in[12];
  unsigned short* wsb = (unsigned short*)d_ws;
  float* out = (float*)d_out;

  (void)in_sizes; (void)n_in; (void)out_size; (void)ws_size;

  int pb = (int)((kWS_ELEMS + 255) / 256);
  emrnn_prepack<<<pb, 256, 0, stream>>>(W_ih0, W_hh0, W_ih1, W_hh1, W_if, wsb);
  emrnn_kernel<<<kBS / 16, 256, 0, stream>>>(x, h0, b_ih0, b_hh0, b_ih1, b_hh1, b_if, wsb, out);
}
